// Stage3PhraseContrastiveModel_65335042507171
// MI455X (gfx1250) — matrix operand order repaired and checked by reading, not compiled, not run on hardware
//
#include <hip/hip_runtime.h>
#include <hip/hip_bf16.h>

typedef _Float16 v8h  __attribute__((ext_vector_type(8)));
typedef _Float16 v16h __attribute__((ext_vector_type(16)));
typedef float    v8f  __attribute__((ext_vector_type(8)));

#define BLOCK 256
#define NWAVE 8
#define ROWS  32

// LDS carve (bytes):
//   sO:    32 x 772 floats  (GEMM f32 output, max width 768 + 4 pad)   = 98816
//   sA:    32 x 776 halves  (f16 activation / A operand, + 8 pad)      = 49664
//   sPart: 32 rows x 8 partials x {sum,sumsq}                          =  2048
//   sStat: 32 rows x {mean, rstd}                                      =   256
#define SO_MAX (ROWS * 772)
#define SA_MAX (ROWS * 776)
#define SMEM_BYTES (SO_MAX * 4 + SA_MAX * 2 + (ROWS * 8 * 2) * 4 + (ROWS * 2) * 4)

// ---------------------------------------------------------------------------
// GEMM stage: out[32 x NOUT] (f32, LDS) = actIn[32 x K] (f16, LDS) @ W^T
// W is row-major [NOUT x K] f16 in global (L2-resident).
//
// Loop order: each wave owns an N-tile and computes BOTH 16-row M-tiles with
// one B-fragment load feeding two v_wmma_f32_16x16x32_f16 — halves L2 weight
// traffic (each weight byte now feeds 2 WMMAs per block).
//
// A fragment (16-bit A 16x32): lane(0..15)=row M, halves 0..15 hold
// K = hi*16 + 0..15   (hi = lane>>4), matching B
// B fragment (16-bit B 32x16): lane(0..15)=col N, halves 0..15 hold
// K = hi*16 + 0..15.
// C/D: vgpr j of lane holds D[M = j + 8*hi][N = lane&15].
// ---------------------------------------------------------------------------
template <int K, int SXA, int SO, int NOUT>
__device__ __forceinline__ void gemm_stage(const _Float16* __restrict__ Wg,
                                           const _Float16* __restrict__ sA,
                                           float* __restrict__ sOut)
{
    const int wave = threadIdx.x >> 5;
    const int lane = threadIdx.x & 31;
    const int lrow = lane & 15;
    const int hi   = lane >> 4;
    constexpr int NT = NOUT / 16;        // N-tiles, distributed across 8 waves

    const _Float16* a0 = sA + lrow * SXA + hi * 16;          // M-tile 0
    const _Float16* a1 = a0 + 16 * SXA;                      // M-tile 1

    for (int nt = wave; nt < NT; nt += NWAVE) {
        const _Float16* brow = Wg + (size_t)(nt * 16 + lrow) * K + hi * 16;
        v8f c0 = {}, c1 = {};
#pragma unroll 4
        for (int k0 = 0; k0 < K; k0 += 32) {
            v16h b = *(const v16h*)(brow + k0);              // 32B contiguous
            v8h l0 = *(const v8h*)(a0 + k0);                 // K = k0+hi*16 .. +7
            v8h h0 = *(const v8h*)(a0 + k0 + 8);             // K = k0+hi*16+8 .. +15
            v16h A0 = __builtin_shufflevector(l0, h0,
                      0,1,2,3,4,5,6,7,8,9,10,11,12,13,14,15);
            v8h l1 = *(const v8h*)(a1 + k0);
            v8h h1 = *(const v8h*)(a1 + k0 + 8);
            v16h A1 = __builtin_shufflevector(l1, h1,
                      0,1,2,3,4,5,6,7,8,9,10,11,12,13,14,15);
            c0 = __builtin_amdgcn_wmma_f32_16x16x32_f16(
                     false, A0, false, b, (short)0, c0, false, false);
            c1 = __builtin_amdgcn_wmma_f32_16x16x32_f16(
                     false, A1, false, b, (short)0, c1, false, false);
        }
        float* od = sOut + nt * 16 + lrow;
        const int rb = hi * 8;
#pragma unroll
        for (int j = 0; j < 8; ++j) {
            od[(rb + j) * SO]      = c0[j];
            od[(rb + 16 + j) * SO] = c1[j];
        }
    }
}

// ---------------------------------------------------------------------------
// LayerNorm (+bias) [+ exact GELU] stage over width W, 32 rows.
// 8 threads per row accumulate partial sum/sumsq; combine; elementwise apply.
// Writes f16 into next A buffer, or NT-streamed f32 to global (final stage) so
// the 402MB output stream does not evict the L2-resident weights.
// ---------------------------------------------------------------------------
template <int W, int SO, int SXO, bool GELU, bool FINAL>
__device__ __forceinline__ void ln_stage(const float* __restrict__ sO,
                                         _Float16* __restrict__ sAout,
                                         float* __restrict__ outg, int nvalid,
                                         const float* __restrict__ bias,
                                         const float* __restrict__ gam,
                                         const float* __restrict__ bet,
                                         float* __restrict__ sPart,
                                         float* __restrict__ sStat)
{
    const int tid = threadIdx.x;
    {
        const int row = tid >> 3, sub = tid & 7;
        float s = 0.f, s2 = 0.f;
        for (int c = sub; c < W; c += 8) {
            float v = sO[row * SO + c] + bias[c];
            s += v; s2 += v * v;
        }
        sPart[(row * 8 + sub) * 2 + 0] = s;
        sPart[(row * 8 + sub) * 2 + 1] = s2;
    }
    __syncthreads();
    if ((tid & 7) == 0) {
        const int row = tid >> 3;
        float S = 0.f, S2 = 0.f;
#pragma unroll
        for (int i = 0; i < 8; ++i) {
            S  += sPart[(row * 8 + i) * 2 + 0];
            S2 += sPart[(row * 8 + i) * 2 + 1];
        }
        const float mean = S * (1.0f / W);
        const float var  = S2 * (1.0f / W) - mean * mean;
        sStat[row * 2 + 0] = mean;
        sStat[row * 2 + 1] = rsqrtf(var + 1e-5f);
    }
    __syncthreads();
    for (int i = tid; i < ROWS * W; i += BLOCK) {
        const int r = i / W, c = i - r * W;
        float v = (sO[r * SO + c] + bias[c] - sStat[r * 2 + 0])
                  * sStat[r * 2 + 1] * gam[c] + bet[c];
        if (GELU) v = 0.5f * v * (1.0f + erff(v * 0.70710678118654752f));
        if constexpr (FINAL) {
            if (r < nvalid)
                __builtin_nontemporal_store(v, &outg[(size_t)r * 768 + c]);
        } else {
            sAout[r * SXO + c] = (_Float16)v;
        }
    }
    __syncthreads();
}

// ---------------------------------------------------------------------------
// Fully fused phrase-encoder kernel: 32 rows per 256-thread block.
// ---------------------------------------------------------------------------
__global__ __launch_bounds__(BLOCK) void phrase_fused_kernel(
    const int* __restrict__ comp_ids, const int* __restrict__ change_ids,
    const float* __restrict__ bbox,
    const float* __restrict__ comp_table, const float* __restrict__ change_table,
    const float* __restrict__ pos_w1, const float* __restrict__ pos_b1,
    const float* __restrict__ pos_g1, const float* __restrict__ pos_bt1,
    const float* __restrict__ pos_w2, const float* __restrict__ pos_b2,
    const _Float16* __restrict__ fus_w1h, const float* __restrict__ fus_b1,
    const float* __restrict__ fus_g1, const float* __restrict__ fus_bt1,
    const _Float16* __restrict__ fus_w2h, const float* __restrict__ fus_b2,
    const float* __restrict__ fus_g2, const float* __restrict__ fus_bt2,
    const _Float16* __restrict__ prj_w1h, const float* __restrict__ prj_b1,
    const float* __restrict__ prj_g1, const float* __restrict__ prj_bt1,
    const _Float16* __restrict__ prj_w2h, const float* __restrict__ prj_b2,
    const float* __restrict__ prj_g2, const float* __restrict__ prj_bt2,
    float* __restrict__ out, int n)
{
    extern __shared__ __align__(16) char smem[];
    float*    sO    = (float*)smem;
    _Float16* sA    = (_Float16*)(smem + SO_MAX * 4);
    float*    sPart = (float*)(smem + SO_MAX * 4 + SA_MAX * 2);
    float*    sStat = sPart + ROWS * 8 * 2;

    const int row0   = blockIdx.x * ROWS;
    const int nvalid = min(ROWS, n - row0);
    const int tid    = threadIdx.x;

    // ---- stage 0: build x[32,160] f16 in sA (stride 168) ----
    // cols 0..63 comp_emb, 64..95 change_emb, 96..159 pos-MLP output
    for (int i = tid; i < ROWS * 96; i += BLOCK) {
        const int r = i / 96, c = i - r * 96;
        int gr = row0 + r; if (gr >= n) gr = n - 1;
        float v = (c < 64) ? comp_table[comp_ids[gr] * 64 + c]
                           : change_table[change_ids[gr] * 32 + (c - 64)];
        sA[r * 168 + c] = (_Float16)v;
    }
    if (tid < ROWS) {
        int gr = row0 + tid; if (gr >= n) gr = n - 1;
        float b4[4];
#pragma unroll
        for (int k = 0; k < 4; ++k)
            b4[k] = fminf(fmaxf(bbox[gr * 4 + k], 0.0f), 1.0f);
        float h[32];
        float s = 0.f, s2 = 0.f;
#pragma unroll
        for (int j = 0; j < 32; ++j) {
            float a = pos_b1[j];
#pragma unroll
            for (int k = 0; k < 4; ++k) a += b4[k] * pos_w1[j * 4 + k];
            h[j] = a; s += a; s2 += a * a;
        }
        const float mean = s * (1.f / 32.f);
        const float rstd = rsqrtf(s2 * (1.f / 32.f) - mean * mean + 1e-5f);
#pragma unroll
        for (int j = 0; j < 32; ++j) {
            float v = (h[j] - mean) * rstd * pos_g1[j] + pos_bt1[j];
            h[j] = 0.5f * v * (1.0f + erff(v * 0.70710678118654752f));
        }
        for (int j = 0; j < 64; ++j) {
            float a = pos_b2[j];
#pragma unroll
            for (int k = 0; k < 32; ++k) a += h[k] * pos_w2[j * 32 + k];
            sA[tid * 168 + 96 + j] = (_Float16)a;
        }
    }
    __syncthreads();

    // ---- fusion: 160->256 (LN,GELU) -> 256->768 (LN,GELU) ----
    gemm_stage<160, 168, 260, 256>(fus_w1h, sA, sO);
    __syncthreads();
    ln_stage<256, 260, 264, true, false>(sO, sA, nullptr, nvalid,
                                         fus_b1, fus_g1, fus_bt1, sPart, sStat);
    gemm_stage<256, 264, 772, 768>(fus_w2h, sA, sO);
    __syncthreads();
    ln_stage<768, 772, 776, true, false>(sO, sA, nullptr, nvalid,
                                         fus_b2, fus_g2, fus_bt2, sPart, sStat);
    // ---- projection: 768->512 (LN,GELU) -> 512->768 (LN) ----
    gemm_stage<768, 776, 516, 512>(prj_w1h, sA, sO);
    __syncthreads();
    ln_stage<512, 516, 520, true, false>(sO, sA, nullptr, nvalid,
                                         prj_b1, prj_g1, prj_bt1, sPart, sStat);
    gemm_stage<512, 520, 772, 768>(prj_w2h, sA, sO);
    __syncthreads();
    ln_stage<768, 772, 0, false, true>(sO, nullptr, out + (size_t)row0 * 768, nvalid,
                                       prj_b2, prj_g2, prj_bt2, sPart, sStat);
}

// ---------------------------------------------------------------------------
// f32 -> f16 weight conversion (into workspace)
// ---------------------------------------------------------------------------
__global__ void cvt_f32_f16(const float* __restrict__ s, _Float16* __restrict__ d, int n)
{
    int i = blockIdx.x * 256 + threadIdx.x;
    if (i < n) d[i] = (_Float16)s[i];
}

extern "C" void kernel_launch(void* const* d_in, const int* in_sizes, int n_in,
                              void* d_out, int out_size, void* d_ws, size_t ws_size,
                              hipStream_t stream)
{
    (void)n_in; (void)out_size; (void)ws_size;
    const int*   comp_ids     = (const int*)d_in[0];
    const int*   change_ids   = (const int*)d_in[1];
    const float* bbox         = (const float*)d_in[2];
    const float* comp_table   = (const float*)d_in[3];
    const float* change_table = (const float*)d_in[4];
    const float* pos_w1  = (const float*)d_in[5];
    const float* pos_b1  = (const float*)d_in[6];
    const float* pos_g1  = (const float*)d_in[7];
    const float* pos_bt1 = (const float*)d_in[8];
    const float* pos_w2  = (const float*)d_in[9];
    const float* pos_b2  = (const float*)d_in[10];
    const float* fus_w1  = (const float*)d_in[11];
    const float* fus_b1  = (const float*)d_in[12];
    const float* fus_g1  = (const float*)d_in[13];
    const float* fus_bt1 = (const float*)d_in[14];
    const float* fus_w2  = (const float*)d_in[15];
    const float* fus_b2  = (const float*)d_in[16];
    const float* fus_g2  = (const float*)d_in[17];
    const float* fus_bt2 = (const float*)d_in[18];
    const float* prj_w1  = (const float*)d_in[19];
    const float* prj_b1  = (const float*)d_in[20];
    const float* prj_g1  = (const float*)d_in[21];
    const float* prj_bt1 = (const float*)d_in[22];
    const float* prj_w2  = (const float*)d_in[23];
    const float* prj_b2  = (const float*)d_in[24];
    const float* prj_g2  = (const float*)d_in[25];
    const float* prj_bt2 = (const float*)d_in[26];
    float* out = (float*)d_out;
    const int n = in_sizes[0];

    // f16 weights in workspace (each offset multiple of 32B for v16h loads)
    _Float16* fus_w1h = (_Float16*)d_ws;              // 256*160
    _Float16* fus_w2h = fus_w1h + 256 * 160;          // 768*256
    _Float16* prj_w1h = fus_w2h + 768 * 256;          // 512*768
    _Float16* prj_w2h = prj_w1h + 512 * 768;          // 768*512

    cvt_f32_f16<<<(256 * 160 + 255) / 256, 256, 0, stream>>>(fus_w1, fus_w1h, 256 * 160);
    cvt_f32_f16<<<(768 * 256 + 255) / 256, 256, 0, stream>>>(fus_w2, fus_w2h, 768 * 256);
    cvt_f32_f16<<<(512 * 768 + 255) / 256, 256, 0, stream>>>(prj_w1, prj_w1h, 512 * 768);
    cvt_f32_f16<<<(768 * 512 + 255) / 256, 256, 0, stream>>>(prj_w2, prj_w2h, 768 * 512);

    hipFuncSetAttribute((const void*)phrase_fused_kernel,
                        hipFuncAttributeMaxDynamicSharedMemorySize, SMEM_BYTES);

    const int blocks = (n + ROWS - 1) / ROWS;
    phrase_fused_kernel<<<blocks, BLOCK, SMEM_BYTES, stream>>>(
        comp_ids, change_ids, bbox, comp_table, change_table,
        pos_w1, pos_b1, pos_g1, pos_bt1, pos_w2, pos_b2,
        fus_w1h, fus_b1, fus_g1, fus_bt1,
        fus_w2h, fus_b2, fus_g2, fus_bt2,
        prj_w1h, prj_b1, prj_g1, prj_bt1,
        prj_w2h, prj_b2, prj_g2, prj_bt2,
        out, n);
}
